// TemporalGNN_35424890257637
// MI455X (gfx1250) — matrix operand order repaired and checked by reading, not compiled, not run on hardware
//
#include <hip/hip_runtime.h>

typedef __attribute__((ext_vector_type(16))) _Float16 v16h;
typedef __attribute__((ext_vector_type(8)))  _Float16 v8h;
typedef __attribute__((ext_vector_type(8)))  float    v8f;
typedef unsigned int __attribute__((ext_vector_type(4))) u32x4;
typedef int __attribute__((ext_vector_type(8)))  i32x8;
typedef int __attribute__((ext_vector_type(4)))  i32x4;

#if defined(__has_builtin)
#if __has_builtin(__builtin_amdgcn_tensor_load_to_lds) && \
    __has_builtin(__builtin_amdgcn_s_wait_tensorcnt)
#define USE_TDM 1
#endif
#endif
#ifndef USE_TDM
#define USE_TDM 0
#endif

namespace {
constexpr int T = 12, N = 10000, E = 320000;
constexpr int NH = N * 128;
constexpr int NROW_P = 10112;       // N padded up to whole 128-row tiles
constexpr int NHP = NROW_P * 128;
constexpr int LSTR = 56;            // LDS halfs/row: 64B data + 48B pad (bank spread)
}

// ---- graph normalization --------------------------------------------------
__global__ void deg_kernel(const int* __restrict__ src, float* __restrict__ deg, int e_cnt) {
  int e = blockIdx.x * blockDim.x + threadIdx.x;
  if (e < e_cnt) unsafeAtomicAdd(&deg[src[e]], 1.0f);
}

__global__ void dinv_kernel(const float* __restrict__ deg, float* __restrict__ dinv, int n) {
  int i = blockIdx.x * blockDim.x + threadIdx.x;
  if (i < n) dinv[i] = deg[i] > 0.0f ? rsqrtf(deg[i]) : 0.0f;
}

__global__ void norm_kernel(const int* __restrict__ src, const int* __restrict__ dst,
                            const float* __restrict__ dinv, float* __restrict__ nrm, int e_cnt) {
  int e = blockIdx.x * blockDim.x + threadIdx.x;
  if (e < e_cnt) nrm[e] = -dinv[src[e]] * dinv[dst[e]];
}

// ---- weight pack: f32 [3,K,H] x2 -> f16 [6 segs][col][k] (transposed) -----
__global__ void packw_kernel(const float* __restrict__ Wx, const float* __restrict__ Wh,
                             _Float16* __restrict__ WT) {
  int i = blockIdx.x * blockDim.x + threadIdx.x;
  if (i >= 6 * 128 * 128) return;
  int seg = i >> 14, rem = i & 16383, c = rem >> 7, k = rem & 127;
  const float* s = (seg < 3) ? Wx : Wh;
  WT[i] = (_Float16)s[((seg % 3) << 14) + (k << 7) + c];
}

// ---- f32 -> padded f16 mirrors -------------------------------------------
__global__ void convx_kernel(const float* __restrict__ x, _Float16* __restrict__ xh) {
  size_t i = (size_t)blockIdx.x * blockDim.x + threadIdx.x;
  if (i >= (size_t)T * NHP) return;
  size_t tt = i / NHP, r = i % NHP;
  xh[i] = (r < NH) ? (_Float16)x[tt * NH + r] : (_Float16)0.0f;
}

__global__ void cvt_kernel(const float* __restrict__ in, _Float16* __restrict__ out) {
  int i = blockIdx.x * blockDim.x + threadIdx.x;
  if (i >= NHP) return;
  out[i] = (i < NH) ? (_Float16)in[i] : (_Float16)0.0f;
}

// ---- Chebyshev combine: out16 = (f16)(2*y - x), padded --------------------
__global__ void axpy2h_kernel(const float* __restrict__ y, const float* __restrict__ x,
                              _Float16* __restrict__ out) {
  int i = blockIdx.x * blockDim.x + threadIdx.x;
  if (i >= NHP) return;
  out[i] = (i < NH) ? (_Float16)(2.0f * y[i] - x[i]) : (_Float16)0.0f;
}

// ---- sparse hop: y[dst] += norm * x[src], one wave32 per edge -------------
__global__ void prop_kernel(const float* __restrict__ x, const int* __restrict__ src,
                            const int* __restrict__ dst, const float* __restrict__ nrm,
                            float* __restrict__ y, int e_cnt) {
  int e = blockIdx.x * 8 + (threadIdx.x >> 5);
  if (e >= e_cnt) return;
  int lane = threadIdx.x & 31;
  int s = src[e], d = dst[e];
  float wv = nrm[e];
  float4 v = ((const float4*)(x + (size_t)s * 128))[lane];
  float* yp = y + (size_t)d * 128 + lane * 4;
  unsafeAtomicAdd(yp + 0, wv * v.x);
  unsafeAtomicAdd(yp + 1, wv * v.y);
  unsafeAtomicAdd(yp + 2, wv * v.z);
  unsafeAtomicAdd(yp + 3, wv * v.w);
}

// ---- fused 6-segment WMMA GEMM (f16 inputs, TDM-staged LDS tiles) ---------
__global__ __launch_bounds__(256) void gemm6h_kernel(
    const _Float16* __restrict__ A0, const _Float16* __restrict__ A1,
    const _Float16* __restrict__ A2, const _Float16* __restrict__ A3,
    const _Float16* __restrict__ A4, const _Float16* __restrict__ A5,
    const _Float16* __restrict__ WTa, float* __restrict__ Ca,
    const _Float16* __restrict__ WTb, float* __restrict__ Cb, int nrows) {
  __shared__ alignas(128) _Float16 As[2][128 * LSTR];
  __shared__ alignas(128) _Float16 Bs[2][128 * LSTR];
  const _Float16* WT = blockIdx.y ? WTb : WTa;
  float* C = blockIdx.y ? Cb : Ca;
  const _Float16* Asegs[6] = {A0, A1, A2, A3, A4, A5};
  const int tid = threadIdx.x;
  const int row0 = blockIdx.x * 128;
  const int w = tid >> 5, lane = tid & 31;
  const int l16 = lane & 15;
  const int sel8 = (lane >> 4) << 3;   // A-frag half-select: 0 (lo lanes) / 8 (hi)
  v8f acc[8] = {};

#if USE_TDM
  const uint32_t asb[2] = {(uint32_t)(uintptr_t)&As[0][0], (uint32_t)(uintptr_t)&As[1][0]};
  const uint32_t bsb[2] = {(uint32_t)(uintptr_t)&Bs[0][0], (uint32_t)(uintptr_t)&Bs[1][0]};
  // D#: 2D tile, 32 x 128 elements of 2B, row stride 128 elems in memory;
  // LDS pad: after every 16 DWORDs (64B row) insert 12 DWORDs (48B) -> 112B = LSTR rows.
  auto tdm_issue = [&](const _Float16* gsrc, uint32_t lds_addr) {
    uint64_t ga = (uint64_t)(uintptr_t)gsrc;
    u32x4 g0;
    g0[0] = 1u;                                                   // count=1
    g0[1] = lds_addr;                                             // lds_addr
    g0[2] = (uint32_t)ga;                                         // global_addr lo
    g0[3] = (uint32_t)((ga >> 32) & 0x01FFFFFFu) | (2u << 30);    // hi | type=2
    i32x8 g1;
    g1[0] = (int)((1u << 16) | (1u << 20) | (3u << 22) | (11u << 25));
    g1[1] = (int)(32u << 16);    // tensor_dim0 = 32
    g1[2] = (int)(128u << 16);   // tensor_dim1 = 128
    g1[3] = (int)(32u << 16);    // tile_dim0 = 32
    g1[4] = 128;                 // tile_dim1 = 128, tile_dim2 = 0
    g1[5] = 128;                 // tensor_dim0_stride = 128
    g1[6] = 0;
    g1[7] = 0;
    i32x4 z4 = {0, 0, 0, 0};
    i32x8 z8 = {0, 0, 0, 0, 0, 0, 0, 0};
    __builtin_amdgcn_tensor_load_to_lds(g0, g1, z4, z4, z8, 0);
  };
  if (tid == 0) {                 // prologue: tiles for iteration 0 into buf 0
    tdm_issue(A0 + (size_t)row0 * 128, asb[0]);
    tdm_issue(WT, bsb[0]);
  }
#endif

  #pragma unroll
  for (int it = 0; it < 24; ++it) {
    const int buf = it & 1;
#if USE_TDM
    if (tid == 0) {
      if (it + 1 < 24) {
        const int s2 = (it + 1) >> 2, k2 = (it + 1) & 3;
        tdm_issue(Asegs[s2] + (size_t)row0 * 128 + k2 * 32, asb[(it + 1) & 1]);
        tdm_issue(WT + ((size_t)s2 << 14) + k2 * 32, bsb[(it + 1) & 1]);
        __builtin_amdgcn_s_wait_tensorcnt(2);   // tile `it` has landed
      } else {
        __builtin_amdgcn_s_wait_tensorcnt(0);
      }
    }
    __syncthreads();
#else
    const int seg = it >> 2, k0 = (it & 3) << 5;
    __syncthreads();
    {  // manual staging fallback: 16B chunks
      const _Float16* Aseg = Asegs[seg];
      const _Float16* Wseg = WT + ((size_t)seg << 14) + k0;
      #pragma unroll
      for (int c = tid; c < 512; c += 256) {
        int r = c >> 2, p = (c & 3) << 3;
        *(v8h*)(&As[buf][r * LSTR + p]) =
            *(const v8h*)(Aseg + (size_t)(row0 + r) * 128 + k0 + p);
        *(v8h*)(&Bs[buf][r * LSTR + p]) =
            *(const v8h*)(Wseg + (size_t)r * 128 + p);
      }
    }
    __syncthreads();
#endif
    // A fragment (16x32): lanes 0-15 need k0-7|k16-23, lanes 16-31 k8-15|k24-31
    const _Float16* ap = &As[buf][(w * 16 + l16) * LSTR];
    v8h a0 = *(const v8h*)(ap + sel8);
    v8h a1 = *(const v8h*)(ap + 16 + sel8);
    v16h a = __builtin_shufflevector(a0, a1,
        0,1,2,3,4,5,6,7,8,9,10,11,12,13,14,15);
    v16h bf[8];
    #pragma unroll
    for (int j = 0; j < 8; ++j) {  // B frag: lanes 0-15 k0-7|k16-23, lanes 16-31 k8-15|k24-31
      const _Float16* bp = &Bs[buf][(j * 16 + l16) * LSTR + sel8];
      v8h b0 = *(const v8h*)(bp);
      v8h b1 = *(const v8h*)(bp + 16);
      bf[j] = __builtin_shufflevector(b0, b1,
          0,1,2,3,4,5,6,7,8,9,10,11,12,13,14,15);
    }
    #pragma unroll
    for (int j = 0; j < 8; ++j)
      acc[j] = __builtin_amdgcn_wmma_f32_16x16x32_f16(false, a, false, bf[j],
                                                      (short)0, acc[j], false, false);
    __syncthreads();
  }

  #pragma unroll
  for (int j = 0; j < 8; ++j) {
    int col = j * 16 + l16;
    #pragma unroll
    for (int i = 0; i < 8; ++i) {
      int m = w * 16 + ((lane >> 4) << 3) + i;   // VGPR i: M = i (+8 for hi lanes)
      int grow = row0 + m;
      if (grow < nrows) C[(size_t)grow * 128 + col] = acc[j][i];
    }
  }
}

// ---- GRU elementwise (write padded f16 mirrors for GEMM operands) ---------
__global__ void zr_kernel(const float* __restrict__ Zp, const float* __restrict__ Rp,
                          const float* __restrict__ bxz, const float* __restrict__ bhz,
                          const float* __restrict__ bxr, const float* __restrict__ bhr,
                          const float* __restrict__ h, float* __restrict__ z,
                          float* __restrict__ g, _Float16* __restrict__ gh) {
  int i = blockIdx.x * blockDim.x + threadIdx.x;
  if (i >= NHP) return;
  if (i < NH) {
    int c = i & 127;
    float zv = 1.0f / (1.0f + expf(-(Zp[i] + bxz[c] + bhz[c])));
    float rv = 1.0f / (1.0f + expf(-(Rp[i] + bxr[c] + bhr[c])));
    float gv = h[i] * rv;
    z[i] = zv;
    g[i] = gv;
    gh[i] = (_Float16)gv;
  } else {
    gh[i] = (_Float16)0.0f;
  }
}

__global__ void newh_kernel(const float* __restrict__ Hp, const float* __restrict__ bxh,
                            const float* __restrict__ bhh, const float* __restrict__ z,
                            float* __restrict__ h, _Float16* __restrict__ hh) {
  int i = blockIdx.x * blockDim.x + threadIdx.x;
  if (i >= NHP) return;
  if (i < NH) {
    int c = i & 127;
    float ht = tanhf(Hp[i] + bxh[c] + bhh[c]);
    float zv = z[i];
    float hn = zv * h[i] + (1.0f - zv) * ht;
    h[i] = hn;
    hh[i] = (_Float16)hn;
  } else {
    hh[i] = (_Float16)0.0f;
  }
}

// ---- output projection H->1, one wave per node ----------------------------
__global__ void outp_kernel(const float* __restrict__ h, const float* __restrict__ Wl,
                            const float* __restrict__ bl, float* __restrict__ out, int n) {
  int node = blockIdx.x * 8 + (threadIdx.x >> 5);
  if (node >= n) return;
  int lane = threadIdx.x & 31;
  float4 hv = ((const float4*)(h + (size_t)node * 128))[lane];
  float4 wv = ((const float4*)Wl)[lane];
  float p = hv.x * wv.x + hv.y * wv.y + hv.z * wv.z + hv.w * wv.w;
  #pragma unroll
  for (int off = 16; off > 0; off >>= 1) p += __shfl_xor(p, off, 32);
  if (lane == 0) out[node] = p + bl[0];
}

// ---------------------------------------------------------------------------
extern "C" void kernel_launch(void* const* d_in, const int* in_sizes, int n_in,
                              void* d_out, int out_size, void* d_ws, size_t ws_size,
                              hipStream_t stream) {
  const float* x_seq = (const float*)d_in[0];
  const int* ei = (const int*)d_in[1];
  const float* Wxz = (const float*)d_in[2];  const float* bxz = (const float*)d_in[3];
  const float* Whz = (const float*)d_in[4];  const float* bhz = (const float*)d_in[5];
  const float* Wxr = (const float*)d_in[6];  const float* bxr = (const float*)d_in[7];
  const float* Whr = (const float*)d_in[8];  const float* bhr = (const float*)d_in[9];
  const float* Wxh = (const float*)d_in[10]; const float* bxh = (const float*)d_in[11];
  const float* Whh = (const float*)d_in[12]; const float* bhh = (const float*)d_in[13];
  const float* Wl = (const float*)d_in[14];  const float* bl = (const float*)d_in[15];
  float* out = (float*)d_out;
  const int* src = ei;
  const int* dst = ei + E;

  char* base = (char*)d_ws;
  size_t off = 0;
  auto alloc = [&](size_t bytes) -> void* {
    void* r = base + off;
    off = (off + bytes + 255) & ~(size_t)255;
    return r;
  };
  float* deg  = (float*)alloc((size_t)N * 4);
  float* dinv = (float*)alloc((size_t)N * 4);
  float* nrm  = (float*)alloc((size_t)E * 4);
  float* Tx1  = (float*)alloc((size_t)NH * 4);
  float* Tx2  = (float*)alloc((size_t)NH * 4);
  float* Th1  = (float*)alloc((size_t)NH * 4);
  float* Th2  = (float*)alloc((size_t)NH * 4);
  float* Tg1  = (float*)alloc((size_t)NH * 4);
  float* Tg2  = (float*)alloc((size_t)NH * 4);
  float* Zpre = (float*)alloc((size_t)NH * 4);
  float* Rpre = (float*)alloc((size_t)NH * 4);
  float* Hpre = (float*)alloc((size_t)NH * 4);
  float* zbuf = (float*)alloc((size_t)NH * 4);
  float* gbuf = (float*)alloc((size_t)NH * 4);
  float* hbuf = (float*)alloc((size_t)NH * 4);
  _Float16* WzT = (_Float16*)alloc((size_t)6 * 128 * 128 * 2);
  _Float16* WrT = (_Float16*)alloc((size_t)6 * 128 * 128 * 2);
  _Float16* WhT = (_Float16*)alloc((size_t)6 * 128 * 128 * 2);
  _Float16* xh   = (_Float16*)alloc((size_t)T * NHP * 2);
  _Float16* Tx1h = (_Float16*)alloc((size_t)NHP * 2);
  _Float16* Tx2h = (_Float16*)alloc((size_t)NHP * 2);
  _Float16* Th1h = (_Float16*)alloc((size_t)NHP * 2);
  _Float16* Th2h = (_Float16*)alloc((size_t)NHP * 2);
  _Float16* Tg1h = (_Float16*)alloc((size_t)NHP * 2);
  _Float16* Tg2h = (_Float16*)alloc((size_t)NHP * 2);
  _Float16* hh   = (_Float16*)alloc((size_t)NHP * 2);
  _Float16* gh   = (_Float16*)alloc((size_t)NHP * 2);

  const int eb = (E + 255) / 256;
  const int nb = (N + 255) / 256;
  const int pb = (E + 7) / 8;          // 1 wave per edge
  const int fp = (NHP + 255) / 256;
  const int ob = (N + 7) / 8;          // 1 wave per node
  const int xb = (int)(((size_t)T * NHP + 255) / 256);
  const dim3 gz((NROW_P + 127) / 128, 2);   // z & r gates in one launch
  const dim3 g1((NROW_P + 127) / 128, 1);

  // one-time setup
  (void)hipMemsetAsync(deg, 0, (size_t)N * 4, stream);
  deg_kernel<<<eb, 256, 0, stream>>>(src, deg, E);
  dinv_kernel<<<nb, 256, 0, stream>>>(deg, dinv, N);
  norm_kernel<<<eb, 256, 0, stream>>>(src, dst, dinv, nrm, E);
  packw_kernel<<<384, 256, 0, stream>>>(Wxz, Whz, WzT);
  packw_kernel<<<384, 256, 0, stream>>>(Wxr, Whr, WrT);
  packw_kernel<<<384, 256, 0, stream>>>(Wxh, Whh, WhT);
  convx_kernel<<<xb, 256, 0, stream>>>(x_seq, xh);
  (void)hipMemsetAsync(hbuf, 0, (size_t)NH * 4, stream);
  (void)hipMemsetAsync(hh, 0, (size_t)NHP * 2, stream);

  for (int t = 0; t < T; ++t) {
    const float* x = x_seq + (size_t)t * NH;
    const _Float16* xht = xh + (size_t)t * NHP;
    // x-side Chebyshev polys
    (void)hipMemsetAsync(Tx1, 0, (size_t)NH * 4, stream);
    prop_kernel<<<pb, 256, 0, stream>>>(x, src, dst, nrm, Tx1, E);
    cvt_kernel<<<fp, 256, 0, stream>>>(Tx1, Tx1h);
    (void)hipMemsetAsync(Tx2, 0, (size_t)NH * 4, stream);
    prop_kernel<<<pb, 256, 0, stream>>>(Tx1, src, dst, nrm, Tx2, E);
    axpy2h_kernel<<<fp, 256, 0, stream>>>(Tx2, x, Tx2h);
    // h-side Chebyshev polys
    (void)hipMemsetAsync(Th1, 0, (size_t)NH * 4, stream);
    prop_kernel<<<pb, 256, 0, stream>>>(hbuf, src, dst, nrm, Th1, E);
    cvt_kernel<<<fp, 256, 0, stream>>>(Th1, Th1h);
    (void)hipMemsetAsync(Th2, 0, (size_t)NH * 4, stream);
    prop_kernel<<<pb, 256, 0, stream>>>(Th1, src, dst, nrm, Th2, E);
    axpy2h_kernel<<<fp, 256, 0, stream>>>(Th2, hbuf, Th2h);
    // z & r pre-activations: K=768 fused WMMA GEMM
    gemm6h_kernel<<<gz, 256, 0, stream>>>(xht, Tx1h, Tx2h, hh, Th1h, Th2h,
                                          WzT, Zpre, WrT, Rpre, N);
    zr_kernel<<<fp, 256, 0, stream>>>(Zpre, Rpre, bxz, bhz, bxr, bhr, hbuf,
                                      zbuf, gbuf, gh);
    // g = h*r side Chebyshev polys
    (void)hipMemsetAsync(Tg1, 0, (size_t)NH * 4, stream);
    prop_kernel<<<pb, 256, 0, stream>>>(gbuf, src, dst, nrm, Tg1, E);
    cvt_kernel<<<fp, 256, 0, stream>>>(Tg1, Tg1h);
    (void)hipMemsetAsync(Tg2, 0, (size_t)NH * 4, stream);
    prop_kernel<<<pb, 256, 0, stream>>>(Tg1, src, dst, nrm, Tg2, E);
    axpy2h_kernel<<<fp, 256, 0, stream>>>(Tg2, gbuf, Tg2h);
    // candidate pre-activation
    gemm6h_kernel<<<g1, 256, 0, stream>>>(xht, Tx1h, Tx2h, gh, Tg1h, Tg2h,
                                          WhT, Hpre, WhT, Hpre, N);
    newh_kernel<<<fp, 256, 0, stream>>>(Hpre, bxh, bhh, zbuf, hbuf, hh);
    outp_kernel<<<ob, 256, 0, stream>>>(hbuf, Wl, bl, out + (size_t)t * N, N);
  }
  (void)in_sizes; (void)n_in; (void)out_size; (void)ws_size;
}
